// Crf_26774826123717
// MI455X (gfx1250) — compile-verified, matrix operand order checked by reading, not run on hardware
//
#include <hip/hip_runtime.h>
#include <hip/hip_bf16.h>

typedef __attribute__((ext_vector_type(16))) _Float16 v16h;
typedef __attribute__((ext_vector_type(8)))  float    v8f;
typedef __attribute__((ext_vector_type(4)))  int      v4i;

typedef __attribute__((address_space(1))) v4i* gv4i_p;   // global 16B vector ptr
typedef __attribute__((address_space(3))) v4i* lv4i_p;   // LDS 16B vector ptr

#define LOG2E 1.4426950408889634f
#define LN2   0.6931471805599453f

constexpr int B_ = 32;
constexpr int L_ = 1024;
constexpr int T_ = 64;
constexpr int CHUNK  = 64;   // pot indices per block in kernel 1
constexpr int NCHUNK = 16;   // 1023 real matrices -> 15 chunks of 64 + 1 of 63
constexpr int STRIDE = 68;   // padded LDS row stride: 272B rows -> 16B aligned,
                             // bank = (4*row + col) & 63 -> conflict-free patterns

#if __has_builtin(__builtin_amdgcn_global_load_async_to_lds_b128) && \
    __has_builtin(__builtin_amdgcn_s_wait_asynccnt)
#define HAS_ASYNC_LDS 1
#else
#define HAS_ASYNC_LDS 0
#endif
#if __has_builtin(__builtin_amdgcn_global_store_async_from_lds_b128) && \
    __has_builtin(__builtin_amdgcn_s_wait_asynccnt)
#define HAS_ASYNC_ST 1
#else
#define HAS_ASYNC_ST 0
#endif

// 16-byte global -> LDS copy (async on CDNA5, ASYNCcnt-tracked; sync fallback)
__device__ __forceinline__ void copy16_g2l(const float* g, float* l) {
#if HAS_ASYNC_LDS
    __builtin_amdgcn_global_load_async_to_lds_b128((gv4i_p)g, (lv4i_p)l, 0, 0);
#else
    *(float4*)l = *(const float4*)g;
#endif
}
// 16-byte LDS -> global copy
__device__ __forceinline__ void copy16_l2g(float* g, const float* l) {
#if HAS_ASYNC_ST
    __builtin_amdgcn_global_store_async_from_lds_b128((gv4i_p)g, (lv4i_p)l, 0, 0);
#else
    *(float4*)g = *(const float4*)l;
#endif
}
template <int N> __device__ __forceinline__ void wait_async() {
#if HAS_ASYNC_LDS || HAS_ASYNC_ST
    __builtin_amdgcn_s_wait_asynccnt(N);
#endif
}

// WMMA fragment helper: element e of a v16h A/B fragment covers
// k_local = (e<8 ? 0 : 16) + (lane<16 ? 0 : 8) + (e&7)  within a 32-wide k-step.
__device__ __forceinline__ int klocal(int e, int grp) {
    return ((e < 8) ? 0 : 16) + grp + (e & 7);
}

// -------- Kernel 1: per (batch, chunk) log-semiring product of up to 64 matrices
// M_l[i,j] = trans[i,j] + logits[b, l+1, j]  (end_states folded into pos L-1).
// Right operand after max-shift is exp(trans[k,j]-tmax[j]) == step-invariant ->
// its WMMA fragments live in VGPRs for the whole chain. The chunk's logits rows
// are async-prefetched into LDS so the 63-step serial chain never touches HBM.
__global__ __launch_bounds__(128)
void crf_chunk_kernel(const float* __restrict__ logits,
                      const float* __restrict__ trans,
                      const float* __restrict__ endst,
                      float* __restrict__ chunkOut)
{
    __shared__ float P[T_ * STRIDE];      // running product (log space)
    __shared__ float TR[T_ * STRIDE];     // transitions
    __shared__ float LG[CHUNK * T_];      // staged logits rows for this chunk
    __shared__ float tmax[T_];
    __shared__ float amArr[T_];
    __shared__ float bmArr[T_];

    const int tid  = threadIdx.x;
    const int lane = tid & 31;
    const int wave = tid >> 5;
    const int b    = blockIdx.x >> 4;     // batch
    const int c    = blockIdx.x & 15;     // chunk
    const int base = c * CHUNK;           // first pot index of this chunk
    const int nmat = (c == NCHUNK - 1) ? (L_ - 1 - base) : CHUNK;  // 63 or 64

    // async-prefetch the chunk's logits block (nmat rows of 64 floats)
    {
        const float* gsrc = logits + ((size_t)b * L_ + base + 1) * T_;
        const int n16 = (nmat * T_) / 4;            // 16B units
        for (int q = tid; q < n16; q += 128)
            copy16_g2l(gsrc + q * 4, &LG[q * 4]);
    }

    // stage transitions into LDS (regular loads; overlaps with async above)
    for (int idx = tid; idx < T_ * T_; idx += 128)
        TR[(idx >> 6) * STRIDE + (idx & 63)] = trans[idx];
    __syncthreads();

    // column max of transitions
    if (tid < T_) {
        float m = -1e30f;
        for (int k = 0; k < T_; ++k) m = fmaxf(m, TR[k * STRIDE + tid]);
        tmax[tid] = m;
    }
    __syncthreads();

    const int grp  = (lane < 16) ? 0 : 8;
    const int nidx = lane & 15;
    const int R0   = wave * 16;           // this wave's 16-row slab of the product
    const int rowA = R0 + nidx;

    // step-invariant B fragments: expB[k][j] = exp(trans[k][j] - tmax[j]) in (0,1]
    v16h bfrag[2][4];
    #pragma unroll
    for (int s = 0; s < 2; ++s)
        #pragma unroll
        for (int ct = 0; ct < 4; ++ct) {
            const int col = ct * 16 + nidx;
            v16h bf;
            #pragma unroll
            for (int e = 0; e < 16; ++e) {
                const int k = 32 * s + klocal(e, grp);
                bf[e] = (_Float16)exp2f((TR[k * STRIDE + col] - tmax[col]) * LOG2E);
            }
            bfrag[s][ct] = bf;
        }

    wait_async<0>();                      // logits block landed in LDS
    __syncthreads();

    // init P = M_{base}
    for (int idx = tid; idx < T_ * T_; idx += 128) {
        const int i = idx >> 6, j = idx & 63;
        P[i * STRIDE + j] = TR[i * STRIDE + j] + LG[j];
    }

    for (int step = 1; step < nmat; ++step) {
        __syncthreads();                         // P current
        const int pos = base + step + 1;         // logits position of this matrix
        if (tid < T_) {
            float v = tmax[tid] + LG[step * T_ + tid];
            if (pos == L_ - 1) v += endst[tid];
            bmArr[tid] = v;                      // column max of right operand
        }
        // row max of P over this wave's slab (2 lanes per row, then xor-16 combine)
        float am = -1e30f;
        {
            const int j0 = (lane < 16) ? 0 : 32;
            const float* prow = &P[rowA * STRIDE + j0];
            #pragma unroll
            for (int j = 0; j < 32; ++j) am = fmaxf(am, prow[j]);
            am = fmaxf(am, __shfl_xor(am, 16));
        }
        if (lane < 16) amArr[R0 + lane] = am;

        // A fragments: exp(P - rowmax) -> f16
        v16h a0, a1;
        #pragma unroll
        for (int e = 0; e < 16; ++e) {
            const int k0 = klocal(e, grp);
            a0[e] = (_Float16)exp2f((P[rowA * STRIDE + k0]      - am) * LOG2E);
            a1[e] = (_Float16)exp2f((P[rowA * STRIDE + 32 + k0] - am) * LOG2E);
        }

        v8f acc[4];
        #pragma unroll
        for (int ct = 0; ct < 4; ++ct) {
            v8f z = {};
            z = __builtin_amdgcn_wmma_f32_16x16x32_f16(false, a0, false, bfrag[0][ct],
                                                       (short)0, z, false, false);
            z = __builtin_amdgcn_wmma_f32_16x16x32_f16(false, a1, false, bfrag[1][ct],
                                                       (short)0, z, false, false);
            acc[ct] = z;
        }
        __syncthreads();                         // all P reads done; bm/am visible
        #pragma unroll
        for (int ct = 0; ct < 4; ++ct) {
            const int col = ct * 16 + nidx;
            #pragma unroll
            for (int r = 0; r < 8; ++r) {
                const int row = R0 + r + grp;
                const float s = fmaxf(acc[ct][r], 1e-30f);
                P[row * STRIDE + col] = amArr[row] + bmArr[col] + log2f(s) * LN2;
            }
        }
    }
    __syncthreads();
    // async store of the de-padded 64x64 product (S_ENDPGM waits all counters)
    float* out = chunkOut + (size_t)(b * NCHUNK + c) * (T_ * T_);
    for (int q = tid; q < (T_ * T_) / 4; q += 128) {
        const int i = q >> 4, j4 = (q & 15) * 4;
        copy16_l2g(out + i * T_ + j4, &P[i * STRIDE + j4]);
    }
    wait_async<0>();
}

// -------- Kernel 2: per batch, fold the 16 chunk products (general log-matmul)
// with double-buffered async chunk staging, then fused final logsumexp.
__global__ __launch_bounds__(128)
void crf_combine_kernel(const float* __restrict__ chunkIn,
                        const float* __restrict__ logits,
                        const float* __restrict__ startst,
                        float* __restrict__ outv)
{
    __shared__ float Q[T_ * STRIDE];
    __shared__ float Bm[2][T_ * STRIDE];  // double-buffered right operand
    __shared__ float amArr[T_];
    __shared__ float cmArr[T_];
    __shared__ float red[128];

    const int tid  = threadIdx.x;
    const int lane = tid & 31;
    const int wave = tid >> 5;
    const int b    = blockIdx.x;
    const int grp  = (lane < 16) ? 0 : 8;
    const int nidx = lane & 15;
    const int R0   = wave * 16;
    const int rowA = R0 + nidx;

    const float* cbase = chunkIn + (size_t)b * NCHUNK * T_ * T_;

    // issue async staging of chunk 1 into Bm[1], then load Q (chunk 0) sync
    {
        const float* src = cbase + (size_t)1 * T_ * T_;
        for (int q = tid; q < (T_ * T_) / 4; q += 128) {
            const int i = q >> 4, j4 = (q & 15) * 4;
            copy16_g2l(src + i * T_ + j4, &Bm[1][i * STRIDE + j4]);
        }
    }
    for (int idx = tid; idx < T_ * T_; idx += 128)
        Q[(idx >> 6) * STRIDE + (idx & 63)] = cbase[idx];

    for (int t = 1; t < NCHUNK; ++t) {
        // issue staging of chunk t+1 into the other buffer (it was consumed at t-1)
        if (t + 1 < NCHUNK) {
            const float* src = cbase + (size_t)(t + 1) * T_ * T_;
            for (int q = tid; q < (T_ * T_) / 4; q += 128) {
                const int i = q >> 4, j4 = (q & 15) * 4;
                copy16_g2l(src + i * T_ + j4, &Bm[(t + 1) & 1][i * STRIDE + j4]);
            }
            wait_async<8>();   // in-order completion: chunk t's 8 loads are done
        } else {
            wait_async<0>();
        }
        __syncthreads();
        const float* Bc = Bm[t & 1];

        if (tid < T_) {                          // column max of right operand
            float m = -1e30f;
            for (int k = 0; k < T_; ++k) m = fmaxf(m, Bc[k * STRIDE + tid]);
            cmArr[tid] = m;
        }
        float am = -1e30f;                       // row max of Q slab
        {
            const int j0 = (lane < 16) ? 0 : 32;
            const float* qrow = &Q[rowA * STRIDE + j0];
            #pragma unroll
            for (int j = 0; j < 32; ++j) am = fmaxf(am, qrow[j]);
            am = fmaxf(am, __shfl_xor(am, 16));
        }
        if (lane < 16) amArr[R0 + lane] = am;
        __syncthreads();                         // cmArr ready

        v16h a0, a1;
        #pragma unroll
        for (int e = 0; e < 16; ++e) {
            const int k0 = klocal(e, grp);
            a0[e] = (_Float16)exp2f((Q[rowA * STRIDE + k0]      - am) * LOG2E);
            a1[e] = (_Float16)exp2f((Q[rowA * STRIDE + 32 + k0] - am) * LOG2E);
        }
        v8f acc[4];
        #pragma unroll
        for (int ct = 0; ct < 4; ++ct) {
            const int col = ct * 16 + nidx;
            v16h b0, b1;
            #pragma unroll
            for (int e = 0; e < 16; ++e) {
                const int k0 = klocal(e, grp);
                b0[e] = (_Float16)exp2f((Bc[k0 * STRIDE + col]        - cmArr[col]) * LOG2E);
                b1[e] = (_Float16)exp2f((Bc[(32 + k0) * STRIDE + col] - cmArr[col]) * LOG2E);
            }
            v8f z = {};
            z = __builtin_amdgcn_wmma_f32_16x16x32_f16(false, a0, false, b0, (short)0, z, false, false);
            z = __builtin_amdgcn_wmma_f32_16x16x32_f16(false, a1, false, b1, (short)0, z, false, false);
            acc[ct] = z;
        }
        __syncthreads();                         // Q reads done
        #pragma unroll
        for (int ct = 0; ct < 4; ++ct) {
            const int col = ct * 16 + nidx;
            #pragma unroll
            for (int r = 0; r < 8; ++r) {
                const int row = R0 + r + grp;
                const float s = fmaxf(acc[ct][r], 1e-30f);
                Q[row * STRIDE + col] = amArr[row] + cmArr[col] + log2f(s) * LN2;
            }
        }
        __syncthreads();                         // Bm[t&1] free for re-staging
    }

    // out[b] = logsumexp_{i,j}( Q[i,j] + logits[b,0,i] + start[i] )
    const float* l0 = logits + (size_t)b * L_ * T_;
    float lm = -1e30f;
    for (int idx = tid; idx < T_ * T_; idx += 128) {
        const int i = idx >> 6, j = idx & 63;
        lm = fmaxf(lm, Q[i * STRIDE + j] + l0[i] + startst[i]);
    }
    red[tid] = lm;
    __syncthreads();
    for (int off = 64; off > 0; off >>= 1) {
        if (tid < off) red[tid] = fmaxf(red[tid], red[tid + off]);
        __syncthreads();
    }
    const float gmax = red[0];
    __syncthreads();
    float ls = 0.f;
    for (int idx = tid; idx < T_ * T_; idx += 128) {
        const int i = idx >> 6, j = idx & 63;
        const float v = Q[i * STRIDE + j] + l0[i] + startst[i];
        ls += exp2f((v - gmax) * LOG2E);
    }
    red[tid] = ls;
    __syncthreads();
    for (int off = 64; off > 0; off >>= 1) {
        if (tid < off) red[tid] += red[tid + off];
        __syncthreads();
    }
    if (tid == 0) outv[b] = gmax + log2f(red[0]) * LN2;
}

extern "C" void kernel_launch(void* const* d_in, const int* in_sizes, int n_in,
                              void* d_out, int out_size, void* d_ws, size_t ws_size,
                              hipStream_t stream) {
    const float* logits = (const float*)d_in[0];   // (32,1024,64)
    const float* trans  = (const float*)d_in[1];   // (64,64)
    const float* startv = (const float*)d_in[2];   // (64,)
    const float* endv   = (const float*)d_in[3];   // (64,)
    // d_in[4] = mask (all ones) -- unused
    float* out    = (float*)d_out;                 // (32,)
    float* chunkW = (float*)d_ws;                  // 32*16*64*64 f32 = 8 MB

    crf_chunk_kernel<<<B_ * NCHUNK, 128, 0, stream>>>(logits, trans, endv, chunkW);
    crf_combine_kernel<<<B_, 128, 0, stream>>>(chunkW, logits, startv, out);
}
